// SatelliteReadingAttention_10651518894745
// MI455X (gfx1250) — matrix operand order repaired and checked by reading, not compiled, not run on hardware
//
#include <hip/hip_runtime.h>
#include <hip/hip_bf16.h>

// ---------------------------------------------------------------------------
// Problem constants (match reference)
// ---------------------------------------------------------------------------
#define BB   2
#define CC   320
#define NF   4096
#define MM   1024
#define SATD 768
#define HH   16
#define DD   64
#define MD   1024
#define LAM  1.0f
#define EPS  1e-5f
#define SCALE 0.125f      // 1/sqrt(64)

typedef __attribute__((ext_vector_type(16))) _Float16 v16h;
typedef __attribute__((ext_vector_type(8)))  _Float16 v8h;
typedef __attribute__((ext_vector_type(8)))  float    v8f;

union V16 { v16h v; v8h h[2]; };

// ---------------------------------------------------------------------------
// CDNA5 async copy: 16B global -> LDS through the async unit (ASYNCcnt)
// ---------------------------------------------------------------------------
__device__ __forceinline__ void async_g2l_b128(unsigned lds_addr, unsigned long long gaddr) {
    asm volatile("global_load_async_to_lds_b128 %0, %1, off"
                 :: "v"(lds_addr), "v"(gaddr) : "memory");
}
__device__ __forceinline__ void wait_async0() {
    asm volatile("s_wait_asynccnt 0x0" ::: "memory");
}

// ---------------------------------------------------------------------------
// fp32 -> f16 conversion (plain)
// ---------------------------------------------------------------------------
__global__ void cvt_kernel(const float* __restrict__ s, _Float16* __restrict__ d, int n) {
    int i = blockIdx.x * 256 + threadIdx.x;
    if (i < n) d[i] = (_Float16)s[i];
}

// ---------------------------------------------------------------------------
// Tiled transpose + convert: in f32 [batch][R][Cc] -> out f16 [batch][Cc][R]
// 32x32 tiles, coalesced on both sides. R, Cc multiples of 32.
// ---------------------------------------------------------------------------
__global__ __launch_bounds__(256) void transcvt_kernel(const float* __restrict__ in,
                                                       _Float16* __restrict__ outp,
                                                       int R, int Cc) {
    __shared__ float tile[32][33];
    int bt = blockIdx.z;
    int r0 = blockIdx.y * 32, c0 = blockIdx.x * 32;
    const float* src = in + (size_t)bt * R * Cc;
    _Float16* dst = outp + (size_t)bt * R * Cc;
    int tx = threadIdx.x & 31, ty = threadIdx.x >> 5;
    #pragma unroll
    for (int i = ty; i < 32; i += 8)
        tile[i][tx] = src[(size_t)(r0 + i) * Cc + c0 + tx];
    __syncthreads();
    #pragma unroll
    for (int i = ty; i < 32; i += 8)
        dst[(size_t)(c0 + i) * R + r0 + tx] = (_Float16)tile[tx][i];
}

// ---------------------------------------------------------------------------
// pos-MLP layer 1: hidden = silu(xy @ w_pos1 + b_pos1), f16 out. One block/row.
// ---------------------------------------------------------------------------
__global__ __launch_bounds__(256) void poshid_kernel(const float* __restrict__ xy,
                                                     const float* __restrict__ w1,
                                                     const float* __restrict__ b1,
                                                     _Float16* __restrict__ hid) {
    int row = blockIdx.x;                       // 0 .. B*NF-1
    float x = xy[(size_t)row * 2 + 0];
    float y = xy[(size_t)row * 2 + 1];
    #pragma unroll
    for (int i = 0; i < 4; i++) {
        int c = threadIdx.x * 4 + i;
        float t = x * w1[c] + y * w1[MD + c] + b1[c];
        float s = t / (1.0f + __expf(-t));      // silu
        hid[(size_t)row * MD + c] = (_Float16)s;
    }
}

// ---------------------------------------------------------------------------
// WMMA "NT" GEMM: C[f32, M x N] = A[f16, M x K] @ Bt[f16, N x K]^T
// No LDS: both operands are contiguous 16B global loads (B pre-transposed).
// 128 threads = 4 waves; wave -> 32x64 (8 WMMA / K-step); block tile 64x128.
// Software-pipelined: double-buffered register fragments so the next K-step's
// loads are in flight while the current step's WMMAs execute (partial
// s_wait_loadcnt instead of full serialization). Requires K % 64 == 0.
// ---------------------------------------------------------------------------
__global__ __launch_bounds__(128) void gemm_nt_kernel(const _Float16* __restrict__ A,
                                                      const _Float16* __restrict__ Bt,
                                                      float* __restrict__ C,
                                                      int Mdim, int Ndim, int Kdim) {
    int tid = threadIdx.x;
    int wave = tid >> 5, lane = tid & 31;
    int grp = lane >> 4, ln = lane & 15;
    int m0 = blockIdx.y * 64 + (wave >> 1) * 32;
    int n0 = blockIdx.x * 128 + (wave & 1) * 64;

    v8f acc[2][4] = {};

    // A-fragment: row = m0(+16)+ln, k(j) = kc + (j>>3)*16 + grp*8 + (j&7)
    const _Float16* a0p = A + (size_t)(m0 + ln) * Kdim + grp * 8;
    const _Float16* a1p = A + (size_t)(m0 + 16 + ln) * Kdim + grp * 8;
    // B-fragment: col = n0 + t*16 + ln, k(j) = kc + (j>>3)*16 + grp*8 + (j&7)
    const _Float16* bbp = Bt + (size_t)(n0 + ln) * Kdim + grp * 8;

    auto loadAB = [&](V16 (&a)[2], V16 (&b)[4], int kc) {
        a[0].h[0] = *(const v8h*)(a0p + kc);
        a[0].h[1] = *(const v8h*)(a0p + kc + 16);
        a[1].h[0] = *(const v8h*)(a1p + kc);
        a[1].h[1] = *(const v8h*)(a1p + kc + 16);
        #pragma unroll
        for (int t = 0; t < 4; t++) {
            const _Float16* bp = bbp + (size_t)t * 16 * Kdim + kc;
            b[t].h[0] = *(const v8h*)(bp);
            b[t].h[1] = *(const v8h*)(bp + 16);
        }
    };
    auto compute = [&](V16 (&a)[2], V16 (&b)[4]) {
        #pragma unroll
        for (int t = 0; t < 4; t++) {
            acc[0][t] = __builtin_amdgcn_wmma_f32_16x16x32_f16(
                false, a[0].v, false, b[t].v, (short)0, acc[0][t], false, false);
            acc[1][t] = __builtin_amdgcn_wmma_f32_16x16x32_f16(
                false, a[1].v, false, b[t].v, (short)0, acc[1][t], false, false);
        }
    };

    V16 aF[2][2]; V16 bF[2][4];
    loadAB(aF[0], bF[0], 0);

    for (int kc = 0; kc < Kdim; kc += 64) {
        // preload kc+32 (always in range: kc <= K-64)
        loadAB(aF[1], bF[1], kc + 32);
        __builtin_prefetch(a0p + kc + 64, 0, 1);      // global_prefetch_b8
        __builtin_prefetch(a1p + kc + 64, 0, 1);
        compute(aF[0], bF[0]);
        if (kc + 64 < Kdim) loadAB(aF[0], bF[0], kc + 64);
        compute(aF[1], bF[1]);
    }

    #pragma unroll
    for (int mi = 0; mi < 2; mi++)
        #pragma unroll
        for (int t = 0; t < 4; t++)
            #pragma unroll
            for (int r = 0; r < 8; r++)
                C[(size_t)(m0 + mi * 16 + grp * 8 + r) * Ndim + n0 + t * 16 + ln] = acc[mi][t][r];
}

// ---------------------------------------------------------------------------
// block reduction helper (sum pair), 256 threads
// ---------------------------------------------------------------------------
__device__ inline void block_reduce2(float& a, float& b, float* sa, float* sb) {
    int tid = threadIdx.x;
    sa[tid] = a; sb[tid] = b;
    __syncthreads();
    for (int s = 128; s > 0; s >>= 1) {
        if (tid < s) { sa[tid] += sa[tid + s]; sb[tid] += sb[tid + s]; }
        __syncthreads();
    }
    a = sa[0]; b = sb[0];
    __syncthreads();
}

// ---------------------------------------------------------------------------
// q_embed = LN( LN(featpre + b_fr)*g_fr+be_fr  +  pos + b_pos2 ) -> f16
// ---------------------------------------------------------------------------
__global__ __launch_bounds__(256) void ln_merge_kernel(
        const float* __restrict__ featpre, const float* __restrict__ bfr,
        const float* __restrict__ gfr, const float* __restrict__ befr,
        const float* __restrict__ pos, const float* __restrict__ bpos2,
        const float* __restrict__ gqn, const float* __restrict__ beqn,
        _Float16* __restrict__ qemb) {
    __shared__ float sa[256], sb[256];
    int row = blockIdx.x, tid = threadIdx.x;
    float v[4], s = 0.f, ss = 0.f;
    #pragma unroll
    for (int i = 0; i < 4; i++) {
        int c = tid * 4 + i;
        float x = featpre[(size_t)row * MD + c] + bfr[c];
        v[i] = x; s += x; ss += x * x;
    }
    block_reduce2(s, ss, sa, sb);
    float mu = s * (1.0f / MD);
    float rstd = rsqrtf(ss * (1.0f / MD) - mu * mu + EPS);

    float t[4]; s = 0.f; ss = 0.f;
    #pragma unroll
    for (int i = 0; i < 4; i++) {
        int c = tid * 4 + i;
        float f = (v[i] - mu) * rstd * gfr[c] + befr[c];
        float x = f + pos[(size_t)row * MD + c] + bpos2[c];
        t[i] = x; s += x; ss += x * x;
    }
    block_reduce2(s, ss, sa, sb);
    float mu2 = s * (1.0f / MD);
    float rstd2 = rsqrtf(ss * (1.0f / MD) - mu2 * mu2 + EPS);
    #pragma unroll
    for (int i = 0; i < 4; i++) {
        int c = tid * 4 + i;
        qemb[(size_t)row * MD + c] = (_Float16)((t[i] - mu2) * rstd2 * gqn[c] + beqn[c]);
    }
}

// ---------------------------------------------------------------------------
// sat = LN(pre + bias)*g + be -> f16
// ---------------------------------------------------------------------------
__global__ __launch_bounds__(256) void ln_kernel(const float* __restrict__ pre,
                                                 const float* __restrict__ bias,
                                                 const float* __restrict__ g,
                                                 const float* __restrict__ be,
                                                 _Float16* __restrict__ outp) {
    __shared__ float sa[256], sb[256];
    int row = blockIdx.x, tid = threadIdx.x;
    float v[4], s = 0.f, ss = 0.f;
    #pragma unroll
    for (int i = 0; i < 4; i++) {
        int c = tid * 4 + i;
        float x = pre[(size_t)row * MD + c] + bias[c];
        v[i] = x; s += x; ss += x * x;
    }
    block_reduce2(s, ss, sa, sb);
    float mu = s * (1.0f / MD);
    float rstd = rsqrtf(ss * (1.0f / MD) - mu * mu + EPS);
    #pragma unroll
    for (int i = 0; i < 4; i++) {
        int c = tid * 4 + i;
        outp[(size_t)row * MD + c] = (_Float16)((v[i] - mu) * rstd * g[c] + be[c]);
    }
}

// ---------------------------------------------------------------------------
// RoPE + bias + head reorder: proj f32 [B*Ntok, MD] -> out f16 [B,H,Ntok,D]
// ---------------------------------------------------------------------------
__global__ void rope_kernel(const float* __restrict__ proj, const float* __restrict__ bias,
                            const float* __restrict__ xy, _Float16* __restrict__ outp,
                            int Ntok) {
    size_t gid = (size_t)blockIdx.x * 256 + threadIdx.x;
    if (gid >= (size_t)BB * Ntok * MD) return;
    int col = (int)(gid & (MD - 1));
    size_t rowg = gid >> 10;                 // b*Ntok + n
    int b = (int)(rowg / Ntok), n = (int)(rowg % Ntok);
    int hh = col >> 6, d = col & 63;
    const float* pr = proj + rowg * MD;
    float o;
    if (d < 32) {
        int seg = d >> 4;                    // 0 -> x, 1 -> y
        int dd = d & 15;
        int i = dd & 7;
        float coord = xy[rowg * 2 + seg];
        float freq = __powf(10000.0f, -(float)i / 8.0f);
        float ang = coord * freq;
        float c = __cosf(ang), sn = __sinf(ang);
        int base = hh * 64 + seg * 16;
        float a  = pr[base + i]     + bias[base + i];
        float b2 = pr[base + 8 + i] + bias[base + 8 + i];
        o = (dd < 8) ? (a * c - b2 * sn) : (a * sn + b2 * c);
    } else {
        o = pr[col] + bias[col];
    }
    outp[(((size_t)b * HH + hh) * Ntok + n) * DD + d] = (_Float16)o;
}

// ---------------------------------------------------------------------------
// v heads + transpose: proj f32 [B*M, MD] + bias -> Vt f16 [B,H,D,M]
// ---------------------------------------------------------------------------
__global__ __launch_bounds__(256) void vtrans_kernel(const float* __restrict__ proj,
                                                     const float* __restrict__ bias,
                                                     _Float16* __restrict__ vt) {
    __shared__ float tile[32][33];
    int bh = blockIdx.z;                     // b*H + h
    int b = bh >> 4, h = bh & 15;
    int m0 = blockIdx.y * 32, d0 = blockIdx.x * 32;
    int tx = threadIdx.x & 31, ty = threadIdx.x >> 5;
    #pragma unroll
    for (int i = ty; i < 32; i += 8)
        tile[i][tx] = proj[(size_t)(b * MM + m0 + i) * MD + h * 64 + d0 + tx] + bias[h * 64 + d0 + tx];
    __syncthreads();
    #pragma unroll
    for (int i = ty; i < 32; i += 8)
        vt[((size_t)bh * DD + d0 + i) * MM + m0 + tx] = (_Float16)tile[tx][i];
}

// ---------------------------------------------------------------------------
// Fused flash attention with geometric bias.
// Block = 128 threads = 4 waves; wave handles one 16-query tile of one (b,h).
// K chunk [32 keys][64 d] and Vt chunk [64 d][32 keys] staged via
// global_load_async_to_lds_b128 (ASYNCcnt). Online softmax per row.
// logits = (Q.K^T)/8 - |q_xy - k_xy|^2 ; out f32 [B, NF, MD].
// ---------------------------------------------------------------------------
__global__ __launch_bounds__(128) void attn_kernel(const _Float16* __restrict__ Q,
                                                   const _Float16* __restrict__ K,
                                                   const _Float16* __restrict__ Vt,
                                                   const float* __restrict__ qxyg,
                                                   const float* __restrict__ kxyg,
                                                   float* __restrict__ outp) {
    __shared__ __align__(16) _Float16 sK[32 * 64];     // [key][d]
    __shared__ __align__(16) _Float16 sVt[64 * 32];    // [d][key]
    __shared__ float sxy[64];                          // [key][2]
    __shared__ __align__(16) _Float16 sP[4][16 * 32];  // per-wave P tile

    int tid = threadIdx.x;
    int wave = tid >> 5, lane = tid & 31;
    int grp = lane >> 4, ln = lane & 15;
    int b = blockIdx.z, h = blockIdx.y;
    int n0 = (blockIdx.x * 4 + wave) * 16;

    // ---- load Q A-fragments (rows = queries, K-dim = d = 64 -> two frags)
    const _Float16* qb = Q + ((size_t)(b * HH + h) * NF + n0) * DD;
    V16 qa0, qa1;
    {
        const _Float16* p0 = qb + ln * DD + 0  + grp * 8;
        const _Float16* p1 = qb + ln * DD + 32 + grp * 8;
        qa0.h[0] = *(const v8h*)(p0); qa0.h[1] = *(const v8h*)(p0 + 16);
        qa1.h[0] = *(const v8h*)(p1); qa1.h[1] = *(const v8h*)(p1 + 16);
    }
    // ---- per-lane query coordinates for its 8 accumulator rows
    float qx[8], qy[8];
    #pragma unroll
    for (int i = 0; i < 8; i++) {
        size_t r = (size_t)b * NF + n0 + grp * 8 + i;
        qx[i] = qxyg[r * 2 + 0];
        qy[i] = qxyg[r * 2 + 1];
    }

    float mrun[8], lrun[8];
    v8f o[4] = {};
    #pragma unroll
    for (int i = 0; i < 8; i++) { mrun[i] = -3.0e38f; lrun[i] = 0.f; }

    _Float16* Pw = sP[wave];
    const size_t kbase  = (size_t)(b * HH + h) * MM * DD;   // K  [key][d]
    const size_t vtbase = (size_t)(b * HH + h) * DD * MM;   // Vt [d][key]

    for (int ch = 0; ch < MM / 32; ch++) {
        __syncthreads();
        // ---- async-stage K chunk and Vt chunk into LDS (16B per lane-op)
        // K:  32 rows x 128B  = 256 x b128 ; Vt: 64 rows x 64B = 256 x b128
        #pragma unroll
        for (int it = 0; it < 4; it++) {
            int idx = tid + it * 128;
            if (idx < 256) {
                int row = idx >> 3, q = idx & 7;           // 8 x 16B per K row
                unsigned lds = (unsigned)(uintptr_t)&sK[row * 64 + q * 8];
                unsigned long long ga = (unsigned long long)(uintptr_t)
                    (K + kbase + (size_t)(ch * 32 + row) * DD + q * 8);
                async_g2l_b128(lds, ga);
            } else {
                int i2 = idx - 256;
                int d = i2 >> 2, q = i2 & 3;               // 4 x 16B per Vt row
                unsigned lds = (unsigned)(uintptr_t)&sVt[d * 32 + q * 8];
                unsigned long long ga = (unsigned long long)(uintptr_t)
                    (Vt + vtbase + (size_t)d * MM + ch * 32 + q * 8);
                async_g2l_b128(lds, ga);
            }
        }
        if (tid < 64) sxy[tid] = kxyg[((size_t)b * MM + ch * 32) * 2 + tid];
        wait_async0();
        __syncthreads();

        // ---- S = Q @ K^T : two 16x16 key tiles, contraction over d (2 frags)
        v8f s0 = {}, s1 = {};
        #pragma unroll
        for (int kc = 0; kc < 2; kc++) {
            V16 bk0, bk1;
            const _Float16* b0 = &sK[(0  + ln) * DD + kc * 32 + grp * 8];
            const _Float16* b1 = &sK[(16 + ln) * DD + kc * 32 + grp * 8];
            bk0.h[0] = *(const v8h*)(b0); bk0.h[1] = *(const v8h*)(b0 + 16);
            bk1.h[0] = *(const v8h*)(b1); bk1.h[1] = *(const v8h*)(b1 + 16);
            const v16h& aq = kc ? qa1.v : qa0.v;
            s0 = __builtin_amdgcn_wmma_f32_16x16x32_f16(false, aq, false, bk0.v, (short)0, s0, false, false);
            s1 = __builtin_amdgcn_wmma_f32_16x16x32_f16(false, aq, false, bk1.v, (short)0, s1, false, false);
        }

        // ---- online softmax with geometric bias
        float kx0 = sxy[2 * ln + 0],        ky0 = sxy[2 * ln + 1];
        float kx1 = sxy[2 * (16 + ln) + 0], ky1 = sxy[2 * (16 + ln) + 1];
        #pragma unroll
        for (int i = 0; i < 8; i++) {
            float dx0 = qx[i] - kx0, dy0 = qy[i] - ky0;
            float dx1 = qx[i] - kx1, dy1 = qy[i] - ky1;
            float l0 = s0[i] * SCALE - LAM * (dx0 * dx0 + dy0 * dy0);
            float l1 = s1[i] * SCALE - LAM * (dx1 * dx1 + dy1 * dy1);
            float cm = fmaxf(l0, l1);
            #pragma unroll
            for (int msk = 1; msk < 16; msk <<= 1) cm = fmaxf(cm, __shfl_xor(cm, msk, 32));
            float mn = fmaxf(mrun[i], cm);
            float alpha = __expf(mrun[i] - mn);
            float p0 = __expf(l0 - mn), p1 = __expf(l1 - mn);
            float rs = p0 + p1;
            #pragma unroll
            for (int msk = 1; msk < 16; msk <<= 1) rs += __shfl_xor(rs, msk, 32);
            lrun[i] = lrun[i] * alpha + rs;
            mrun[i] = mn;
            o[0][i] *= alpha; o[1][i] *= alpha; o[2][i] *= alpha; o[3][i] *= alpha;
            int row = grp * 8 + i;
            Pw[row * 32 + ln]      = (_Float16)p0;
            Pw[row * 32 + 16 + ln] = (_Float16)p1;
        }
        // intra-wave LDS RAW (LDS pipe in-order; make the dependency explicit)
        asm volatile("s_wait_dscnt 0" ::: "memory");

        // ---- P A-fragment: rows = queries, K-dim = 32 keys of this chunk
        V16 pa;
        const _Float16* pp = Pw + ln * 32 + grp * 8;
        pa.h[0] = *(const v8h*)(pp);
        pa.h[1] = *(const v8h*)(pp + 16);

        // ---- O += P @ V : four 16x16 d-tiles, contraction over 32 keys
        #pragma unroll
        for (int dt = 0; dt < 4; dt++) {
            V16 bv_;
            const _Float16* vb = &sVt[(dt * 16 + ln) * 32 + grp * 8];
            bv_.h[0] = *(const v8h*)(vb);
            bv_.h[1] = *(const v8h*)(vb + 16);
            o[dt] = __builtin_amdgcn_wmma_f32_16x16x32_f16(false, pa.v, false, bv_.v, (short)0, o[dt], false, false);
        }
    }

    // ---- normalize and store: out[b, n, h*64 + d] f32
    #pragma unroll
    for (int i = 0; i < 8; i++) {
        float inv = 1.0f / lrun[i];
        int row = grp * 8 + i;
        size_t obase = ((size_t)b * NF + n0 + row) * MD + h * 64;
        #pragma unroll
        for (int dt = 0; dt < 4; dt++)
            outp[obase + dt * 16 + ln] = o[dt][i] * inv;
    }
}

// ---------------------------------------------------------------------------
// Host orchestration
// ---------------------------------------------------------------------------
extern "C" void kernel_launch(void* const* d_in, const int* in_sizes, int n_in,
                              void* d_out, int out_size, void* d_ws, size_t ws_size,
                              hipStream_t stream) {
    (void)in_sizes; (void)n_in; (void)out_size; (void)ws_size;

    const float* front_feat = (const float*)d_in[0];
    const float* front_xy   = (const float*)d_in[1];
    const float* sat_tokens = (const float*)d_in[2];
    const float* sat_xy     = (const float*)d_in[3];
    const float* w_pos1     = (const float*)d_in[4];
    const float* b_pos1     = (const float*)d_in[5];
    const float* w_pos2     = (const float*)d_in[6];
    const float* b_pos2     = (const float*)d_in[7];
    const float* w_fr       = (const float*)d_in[8];
    const float* b_fr       = (const float*)d_in[9];
    const float* g_fr       = (const float*)d_in[10];
    const float* be_fr      = (const float*)d_in[11];
    const float* g_qn       = (const float*)d_in[12];
    const float* be_qn      = (const float*)d_in[13];
    const float* w_sa       = (const float*)d_in[14];
    const float* b_sa       = (const float*)d_in[15];
    const float* g_sa       = (const float*)d_in[16];
    const float* be_sa      = (const float*)d_in[17];
    const float* wq         = (const float*)d_in[18];
    const float* bq         = (const float*)d_in[19];
    const float* wk         = (const float*)d_in[20];
    const float* bk         = (const float*)d_in[21];
    const float* wv         = (const float*)d_in[22];
    const float* bv         = (const float*)d_in[23];
    float* outp             = (float*)d_out;

    // ---- workspace bump allocator (deterministic)
    char* ws = (char*)d_ws;
    size_t off = 0;
    auto alloc = [&](size_t bytes) -> char* {
        char* p = ws + off;
        off = (off + bytes + 255) & ~(size_t)255;
        return p;
    };
    const int ROWS_F = BB * NF;   // 8192
    const int ROWS_S = BB * MM;   // 2048

    // weights pre-transposed to [N][K] f16 for LDS-free NT GEMM
    _Float16* wpos2t = (_Float16*)alloc((size_t)MD * MD * 2);
    _Float16* wfrt   = (_Float16*)alloc((size_t)CC * MD * 2);
    _Float16* wsat   = (_Float16*)alloc((size_t)SATD * MD * 2);
    _Float16* wqt    = (_Float16*)alloc((size_t)MD * MD * 2);
    _Float16* wkt    = (_Float16*)alloc((size_t)MD * MD * 2);
    _Float16* wvt    = (_Float16*)alloc((size_t)MD * MD * 2);
    _Float16* hidh   = (_Float16*)alloc((size_t)ROWS_F * MD * 2);  // hidden, reused as q_embed
    _Float16* ffh    = (_Float16*)alloc((size_t)ROWS_F * CC * 2);  // front_flat f16 [B*NF, C]
    _Float16* satinh = (_Float16*)alloc((size_t)ROWS_S * SATD * 2);
    _Float16* sath   = (_Float16*)alloc((size_t)ROWS_S * MD * 2);
    _Float16* qh     = (_Float16*)alloc((size_t)ROWS_F * MD * 2);  // [B,H,NF,D]
    _Float16* kh     = (_Float16*)alloc((size_t)ROWS_S * MD * 2);  // [B,H,M,D]
    _Float16* vth    = (_Float16*)alloc((size_t)ROWS_S * MD * 2);  // Vt [B,H,D,M]
    float*    F1     = (float*)alloc((size_t)ROWS_F * MD * 4);     // featpre -> qproj
    float*    F2     = (float*)alloc((size_t)ROWS_F * MD * 4);     // pos -> satpre -> k/vproj

    // 1) weight transposes (f32 [K,N] -> f16 [N,K]) + activation conversions
    transcvt_kernel<<<dim3(MD / 32, MD / 32, 1),   256, 0, stream>>>(w_pos2, wpos2t, MD,   MD);
    transcvt_kernel<<<dim3(MD / 32, CC / 32, 1),   256, 0, stream>>>(w_fr,   wfrt,   CC,   MD);
    transcvt_kernel<<<dim3(MD / 32, SATD / 32, 1), 256, 0, stream>>>(w_sa,   wsat,   SATD, MD);
    transcvt_kernel<<<dim3(MD / 32, MD / 32, 1),   256, 0, stream>>>(wq,     wqt,    MD,   MD);
    transcvt_kernel<<<dim3(MD / 32, MD / 32, 1),   256, 0, stream>>>(wk,     wkt,    MD,   MD);
    transcvt_kernel<<<dim3(MD / 32, MD / 32, 1),   256, 0, stream>>>(wv,     wvt,    MD,   MD);
    // front_feat [B][C][NF] -> front_flat f16 [B][NF][C]
    transcvt_kernel<<<dim3(NF / 32, CC / 32, BB),  256, 0, stream>>>(front_feat, ffh, CC, NF);
    cvt_kernel<<<(ROWS_S * SATD + 255) / 256, 256, 0, stream>>>(sat_tokens, satinh, ROWS_S * SATD);

    // 2) pos-MLP layer 1
    poshid_kernel<<<ROWS_F, 256, 0, stream>>>(front_xy, w_pos1, b_pos1, hidh);

    // 3) GEMMs: featpre = front_flat @ w_fr ; pos = hidden @ w_pos2
    gemm_nt_kernel<<<dim3(MD / 128, ROWS_F / 64), 128, 0, stream>>>(ffh,  wfrt,   F1, ROWS_F, MD, CC);
    gemm_nt_kernel<<<dim3(MD / 128, ROWS_F / 64), 128, 0, stream>>>(hidh, wpos2t, F2, ROWS_F, MD, MD);

    // 4) q_embed = LN(LN(featpre+b)*g+b + pos+b) -> f16 (reuse hidh)
    ln_merge_kernel<<<ROWS_F, 256, 0, stream>>>(F1, b_fr, g_fr, be_fr, F2, b_pos2, g_qn, be_qn, hidh);

    // 5) sat = LN(sat_tokens @ w_sa + b_sa)
    gemm_nt_kernel<<<dim3(MD / 128, ROWS_S / 64), 128, 0, stream>>>(satinh, wsat, F2, ROWS_S, MD, SATD);
    ln_kernel<<<ROWS_S, 256, 0, stream>>>(F2, b_sa, g_sa, be_sa, sath);

    // 6) q = rope(heads(q_embed @ wq + bq))
    gemm_nt_kernel<<<dim3(MD / 128, ROWS_F / 64), 128, 0, stream>>>(hidh, wqt, F1, ROWS_F, MD, MD);
    rope_kernel<<<(int)(((size_t)ROWS_F * MD + 255) / 256), 256, 0, stream>>>(F1, bq, front_xy, qh, NF);

    // 7) k = rope(heads(sat @ wk + bk))
    gemm_nt_kernel<<<dim3(MD / 128, ROWS_S / 64), 128, 0, stream>>>(sath, wkt, F2, ROWS_S, MD, MD);
    rope_kernel<<<(int)(((size_t)ROWS_S * MD + 255) / 256), 256, 0, stream>>>(F2, bk, sat_xy, kh, MM);

    // 8) Vt = heads(sat @ wv + bv) transposed to [B,H,D,M]
    gemm_nt_kernel<<<dim3(MD / 128, ROWS_S / 64), 128, 0, stream>>>(sath, wvt, F2, ROWS_S, MD, MD);
    vtrans_kernel<<<dim3(DD / 32, MM / 32, BB * HH), 256, 0, stream>>>(F2, bv, vth);

    // 9) fused flash attention with geometric bias -> out f32 [B,NF,MD]
    attn_kernel<<<dim3(NF / 64, HH, BB), 128, 0, stream>>>(qh, kh, vth, front_xy, sat_xy, outp);
}
